// GFCNB_7395933683878
// MI455X (gfx1250) — matrix operand order repaired and checked by reading, not compiled, not run on hardware
//
#include <hip/hip_runtime.h>
#include <algorithm>

typedef __attribute__((ext_vector_type(16))) _Float16 v16h;
typedef __attribute__((ext_vector_type(8)))  _Float16 v8h;
typedef __attribute__((ext_vector_type(8)))  float    v8f;

static inline long long divup_ll(long long a, long long b){ return (a + b - 1) / b; }

// ---------------------------------------------------------------- utilities

__global__ void k_zero_u32(unsigned* __restrict__ p, long long n){
  long long i = (long long)blockIdx.x * blockDim.x + threadIdx.x;
  if (i < n) p[i] = 0u;
}

// degree-1 open B-spline basis on dim-2 pseudo coords -> 4 (idx, w) per edge
__global__ void k_basis(const float* __restrict__ attr, int E, int K,
                        int* __restrict__ idx4, float* __restrict__ w4){
  int e = blockIdx.x * blockDim.x + threadIdx.x;
  if (e >= E) return;
  float p0 = attr[2*e + 0] * (float)(K - 1);
  float p1 = attr[2*e + 1] * (float)(K - 1);
  float lo0 = fminf(fmaxf(floorf(p0), 0.f), (float)(K - 2));
  float lo1 = fminf(fmaxf(floorf(p1), 0.f), (float)(K - 2));
  float f0 = p0 - lo0, f1 = p1 - lo1;
  int i0 = (int)lo0, i1 = (int)lo1;
  #pragma unroll
  for (int j = 0; j < 4; ++j){
    int c0 = j >> 1, c1 = j & 1;
    idx4[4*e + j] = (i0 + c0) + K * (i1 + c1);
    w4[4*e + j] = (c0 ? f0 : 1.f - f0) * (c1 ? f1 : 1.f - f1);
  }
}

// histogram counts (node in-degree, cluster sizes)
__global__ void k_count(const int* __restrict__ ids, int n, float* __restrict__ cnt){
  int i = blockIdx.x * blockDim.x + threadIdx.x;
  if (i < n) atomicAdd(&cnt[ids[i]], 1.f);
}

// per (edge, channel): scatter 4 weighted contributions into s[n][Kt][Cin]
__global__ void k_scatter(const float* __restrict__ x, const int* __restrict__ src,
                          const int* __restrict__ dst, int E, int Cin, int Kt,
                          const int* __restrict__ idx4, const float* __restrict__ w4,
                          float* __restrict__ s){
  long long t = (long long)blockIdx.x * blockDim.x + threadIdx.x;
  if (t >= (long long)E * Cin) return;
  int e = (int)(t / Cin), c = (int)(t % Cin);
  float xv = x[(long long)src[e] * Cin + c];
  long long base = ((long long)dst[e] * Kt) * Cin + c;
  #pragma unroll
  for (int j = 0; j < 4; ++j)
    atomicAdd(&s[base + (long long)idx4[4*e + j] * Cin], w4[4*e + j] * xv);
}

// pack A = [ s/deg | x | 0-pad ] as f16 row-major [Mpad][Kpad]
__global__ void k_pack_A(const float* __restrict__ s, const float* __restrict__ x,
                         const float* __restrict__ deg, _Float16* __restrict__ A,
                         int n, int Cin, int Kt, int Kpad, int Mpad){
  long long t = (long long)blockIdx.x * blockDim.x + threadIdx.x;
  if (t >= (long long)Mpad * Kpad) return;
  int m = (int)(t / Kpad), k = (int)(t % Kpad);
  float v = 0.f;
  if (m < n){
    int KC = Kt * Cin;
    if (k < KC)            v = s[(long long)m * KC + k] / fmaxf(deg[m], 1.f);
    else if (k < KC + Cin) v = x[(long long)m * Cin + (k - KC)];
  }
  A[t] = (_Float16)v;
}

// pack Bt = [ W^T | root^T | 0-pad ] as f16 row-major [Npad][Kpad]; bias padded to f32[Npad]
__global__ void k_pack_W(const float* __restrict__ W, const float* __restrict__ root,
                         const float* __restrict__ bias, _Float16* __restrict__ Bt,
                         float* __restrict__ biasp, int Cout, int Cin, int Kt,
                         int Kpad, int Npad){
  long long t = (long long)blockIdx.x * blockDim.x + threadIdx.x;
  if (t >= (long long)Npad * Kpad) return;
  int co = (int)(t / Kpad), k = (int)(t % Kpad);
  float v = 0.f;
  if (co < Cout){
    int KC = Kt * Cin;
    if (k < KC)            v = W[(long long)k * Cout + co];
    else if (k < KC + Cin) v = root[(long long)(k - KC) * Cout + co];
  }
  Bt[t] = (_Float16)v;
  if (k == 0) biasp[co] = (co < Cout) ? bias[co] : 0.f;
}

// ---------------------------------------------------------------- WMMA GEMM
// One wave32 per 32x32 C macro-tile = 2x2 register-blocked 16x16 WMMA tiles.
// Per K-step: 2 A-fragments + 2 B-fragments (8x b128 loads) feed 4 WMMAs ->
// 2 loads/WMMA (vs 4 unblocked), 2-way register reuse of each fragment.
// A frag (16-bit 16x32): lanes 0-15 row=l hold K 0..15; lanes 16-31 K 16..31.
// B frag (16-bit 32x16): lanes 0-15 col=l hold K 0..15; lanes 16-31 K 16..31.
// C layout: VGPR r, lanes 0-15 -> (M=r, N=l); lanes 16-31 -> (M=r+8, N=l).
__global__ void k_gemm32(const _Float16* __restrict__ A, const _Float16* __restrict__ Bt,
                         const float* __restrict__ biasp, float* __restrict__ out,
                         int M, int Cout, int Kpad, int act){
  const int tm = blockIdx.x << 5;
  const int tn = blockIdx.y << 5;
  const int lane = threadIdx.x;
  const int grp = lane >> 4;
  const int l = lane & 15;
  const _Float16* a0 = A  + (long long)(tm + l) * Kpad + grp * 16;
  const _Float16* a1 = a0 + (long long)16 * Kpad;
  const _Float16* b0 = Bt + (long long)(tn + l) * Kpad + grp * 16;
  const _Float16* b1 = b0 + (long long)16 * Kpad;
  v8f acc00 = {}, acc01 = {}, acc10 = {}, acc11 = {};
  for (int k0 = 0; k0 < Kpad; k0 += 32){
    v8h a0lo = *(const v8h*)(a0);
    v8h a0hi = *(const v8h*)(a0 + 8);
    v8h a1lo = *(const v8h*)(a1);
    v8h a1hi = *(const v8h*)(a1 + 8);
    v16h bf0 = *(const v16h*)(b0);
    v16h bf1 = *(const v16h*)(b1);
    v16h af0, af1;
    #pragma unroll
    for (int i = 0; i < 8; ++i){
      af0[i] = a0lo[i]; af0[i + 8] = a0hi[i];
      af1[i] = a1lo[i]; af1[i + 8] = a1hi[i];
    }
    __builtin_prefetch(a0 + 32, 0, 1);
    __builtin_prefetch(a1 + 32, 0, 1);
    __builtin_prefetch(b0 + 32, 0, 1);
    __builtin_prefetch(b1 + 32, 0, 1);
    acc00 = __builtin_amdgcn_wmma_f32_16x16x32_f16(false, af0, false, bf0, (short)0, acc00, false, false);
    acc01 = __builtin_amdgcn_wmma_f32_16x16x32_f16(false, af0, false, bf1, (short)0, acc01, false, false);
    acc10 = __builtin_amdgcn_wmma_f32_16x16x32_f16(false, af1, false, bf0, (short)0, acc10, false, false);
    acc11 = __builtin_amdgcn_wmma_f32_16x16x32_f16(false, af1, false, bf1, (short)0, acc11, false, false);
    a0 += 32; a1 += 32; b0 += 32; b1 += 32;
  }
  v8f accs[4] = { acc00, acc01, acc10, acc11 };
  #pragma unroll
  for (int tM = 0; tM < 2; ++tM){
    #pragma unroll
    for (int tN = 0; tN < 2; ++tN){
      v8f acc = accs[tM * 2 + tN];
      int n = tn + tN * 16 + l;
      if (n < Cout){
        float bv = biasp[n];
        int mbase = tm + tM * 16 + (grp << 3);
        #pragma unroll
        for (int r = 0; r < 8; ++r){
          int m = mbase + r;
          if (m < M){
            float v = acc[r] + bv;
            if (act == 1)      v = v > 0.f ? v : (expf(v) - 1.f);   // ELU
            else if (act == 2) v = 1.f / (1.f + expf(-v));          // sigmoid
            out[(long long)m * Cout + n] = v;
          }
        }
      }
    }
  }
}

// ---------------------------------------------------------------- batchnorm

__global__ void k_bn_stats(const float* __restrict__ x, int n, int C,
                           float* __restrict__ meanvar){
  __shared__ float sh[512];
  int c = blockIdx.x;
  int tid = threadIdx.x;
  float s = 0.f, s2 = 0.f;
  for (int i = tid; i < n; i += blockDim.x){
    float v = x[(long long)i * C + c];
    s += v; s2 += v * v;
  }
  sh[tid] = s; sh[tid + 256] = s2; __syncthreads();
  for (int o = 128; o > 0; o >>= 1){
    if (tid < o){ sh[tid] += sh[tid + o]; sh[tid + 256] += sh[tid + 256 + o]; }
    __syncthreads();
  }
  if (tid == 0){
    float mu = sh[0] / (float)n;
    meanvar[2*c]     = mu;
    meanvar[2*c + 1] = sh[256] / (float)n - mu * mu;
  }
}

__global__ void k_bn_apply(float* __restrict__ x, int n, int C,
                           const float* __restrict__ meanvar,
                           const float* __restrict__ gamma,
                           const float* __restrict__ beta){
  long long t = (long long)blockIdx.x * blockDim.x + threadIdx.x;
  if (t >= (long long)n * C) return;
  int c = (int)(t % C);
  x[t] = (x[t] - meanvar[2*c]) * rsqrtf(meanvar[2*c + 1] + 1e-5f) * gamma[c] + beta[c];
}

// ---------------------------------------------------------------- max pool / unpool

__device__ __forceinline__ unsigned enc_f(float f){
  unsigned u = __float_as_uint(f);
  return (u & 0x80000000u) ? ~u : (u | 0x80000000u);
}
__device__ __forceinline__ float dec_f(unsigned u){
  return (u & 0x80000000u) ? __uint_as_float(u & 0x7FFFFFFFu) : __uint_as_float(~u);
}

__global__ void k_pool_max(const float* __restrict__ x, const int* __restrict__ cl,
                           int n, int C, unsigned* __restrict__ enc){
  long long t = (long long)blockIdx.x * blockDim.x + threadIdx.x;
  if (t >= (long long)n * C) return;
  int i = (int)(t / C), c = (int)(t % C);
  atomicMax(&enc[(long long)cl[i] * C + c], enc_f(x[t]));
}

__global__ void k_pool_fin(const unsigned* __restrict__ enc, const float* __restrict__ cnt,
                           float* __restrict__ out, int n2, int C){
  long long t = (long long)blockIdx.x * blockDim.x + threadIdx.x;
  if (t >= (long long)n2 * C) return;
  int i = (int)(t / C);
  out[t] = (cnt[i] > 0.f) ? dec_f(enc[t]) : 0.f;
}

__global__ void k_gather(const float* __restrict__ in, const int* __restrict__ cl,
                         float* __restrict__ out, int n, int C){
  long long t = (long long)blockIdx.x * blockDim.x + threadIdx.x;
  if (t >= (long long)n * C) return;
  int i = (int)(t / C), c = (int)(t % C);
  out[t] = in[(long long)cl[i] * C + c];
}

// ---------------------------------------------------------------- launch

extern "C" void kernel_launch(void* const* d_in, const int* in_sizes, int n_in,
                              void* d_out, int out_size, void* d_ws, size_t ws_size,
                              hipStream_t stream){
  if (n_in < 44) return;
  // input order: x, attr1..3, then pytree-flattened params (sorted keys:
  // bn_beta, bn_gamma, c1a{W,bias,root}, c1b, c1c, c2a, c2b, c2c, c3a, c3b,
  // c4a, c4b, out), then ei1..3, cl1, cl2.
  const float* x0      = (const float*)d_in[0];
  const float* attr1   = (const float*)d_in[1];
  const float* attr2   = (const float*)d_in[2];
  const float* attr3   = (const float*)d_in[3];
  const float* bn_beta = (const float*)d_in[4];
  const float* bn_gamma= (const float*)d_in[5];
  const int* ei1 = (const int*)d_in[39];
  const int* ei2 = (const int*)d_in[40];
  const int* ei3 = (const int*)d_in[41];
  const int* cl1 = (const int*)d_in[42];
  const int* cl2 = (const int*)d_in[43];
  const int n1 = in_sizes[0];
  const int E1 = in_sizes[39] / 2;
  const int E2 = in_sizes[40] / 2;
  const int E3 = in_sizes[41] / 2;
  const int n2 = in_sizes[43];

  // bump allocator over workspace
  char* base = (char*)d_ws; size_t off = 0;
  auto alloc = [&](size_t bytes) -> void* {
    off = (off + 255) & ~(size_t)255;
    void* p = base + off; off += bytes; return p;
  };
  int*   idx15 = (int*)  alloc((size_t)E1 * 16);
  float* w15   = (float*)alloc((size_t)E1 * 16);
  int*   idx23 = (int*)  alloc((size_t)E2 * 16);
  float* w23   = (float*)alloc((size_t)E2 * 16);
  int*   idx25 = (int*)  alloc((size_t)E2 * 16);
  float* w25   = (float*)alloc((size_t)E2 * 16);
  int*   idx33 = (int*)  alloc((size_t)E3 * 16);
  float* w33   = (float*)alloc((size_t)E3 * 16);
  float* deg1  = (float*)alloc((size_t)n1 * 4);
  float* deg2  = (float*)alloc((size_t)n2 * 4);
  float* deg3  = (float*)alloc((size_t)n2 * 4);
  float* cnt1  = (float*)alloc((size_t)n2 * 4);
  float* cnt2  = (float*)alloc((size_t)n2 * 4);
  float* meanvar = (float*)alloc(128 * 4);
  float* biasp   = (float*)alloc(256 * 4);
  long long bufE = std::max((long long)n1 * 64, (long long)n2 * 256);
  float* bufX = (float*)alloc((size_t)bufE * 4);
  float* bufY = (float*)alloc((size_t)bufE * 4);
  unsigned* encb = (unsigned*)alloc((size_t)n2 * 256 * 4);

  long long maxS = 0, maxA = 0, maxB = 0;
  auto note = [&](long long nn, int K, int Cin, int Cout){
    long long KC = (long long)K * K * Cin;
    long long Kpad = divup_ll(KC + Cin, 32) * 32;
    maxS = std::max(maxS, nn * KC);
    maxA = std::max(maxA, divup_ll(nn, 32) * 32 * Kpad);
    maxB = std::max(maxB, divup_ll(Cout, 32) * 32 * Kpad);
  };
  note(n1,5,1,32);   note(n1,5,32,64);   note(n1,5,64,64);
  note(n2,3,64,128); note(n2,3,128,128); note(n2,3,128,256);
  note(n2,3,256,256); note(n2,3,256,128);
  note(n2,5,128,64); note(n2,5,64,32);  note(n1,5,32,1);
  float*    sbuf = (float*)   alloc((size_t)maxS * 4);
  _Float16* Abuf = (_Float16*)alloc((size_t)maxA * 2);
  _Float16* Bbuf = (_Float16*)alloc((size_t)maxB * 2);
  (void)ws_size; (void)out_size;

  auto zero = [&](void* p, long long words){
    k_zero_u32<<<(unsigned)divup_ll(words, 256), 256, 0, stream>>>((unsigned*)p, words);
  };

  // graph-level constants (edge attrs / topology are layer-invariant)
  zero(deg1, n1); zero(deg2, n2); zero(deg3, n2); zero(cnt1, n2); zero(cnt2, n2);
  k_basis<<<(unsigned)divup_ll(E1,256),256,0,stream>>>(attr1, E1, 5, idx15, w15);
  k_basis<<<(unsigned)divup_ll(E2,256),256,0,stream>>>(attr2, E2, 3, idx23, w23);
  k_basis<<<(unsigned)divup_ll(E2,256),256,0,stream>>>(attr2, E2, 5, idx25, w25);
  k_basis<<<(unsigned)divup_ll(E3,256),256,0,stream>>>(attr3, E3, 3, idx33, w33);
  k_count<<<(unsigned)divup_ll(E1,256),256,0,stream>>>(ei1 + E1, E1, deg1);
  k_count<<<(unsigned)divup_ll(E2,256),256,0,stream>>>(ei2 + E2, E2, deg2);
  k_count<<<(unsigned)divup_ll(E3,256),256,0,stream>>>(ei3 + E3, E3, deg3);
  k_count<<<(unsigned)divup_ll(n1,256),256,0,stream>>>(cl1, n1, cnt1);
  k_count<<<(unsigned)divup_ll(n2,256),256,0,stream>>>(cl2, n2, cnt2);

  auto conv = [&](const float* xin, float* xout, int nn, const int* ei, int E,
                  const int* idx4, const float* w4, const float* deg,
                  int wbase, int K, int Cin, int Cout, int act){
    const float* W    = (const float*)d_in[wbase];
    const float* bias = (const float*)d_in[wbase + 1];
    const float* root = (const float*)d_in[wbase + 2];
    int Kt = K * K;
    long long KC = (long long)Kt * Cin;
    int Kpad = (int)(divup_ll(KC + Cin, 32) * 32);
    int Mpad = (int)(divup_ll(nn, 32) * 32);
    int Npad = (int)(divup_ll(Cout, 32) * 32);
    zero(sbuf, (long long)nn * KC);
    long long sc = (long long)E * Cin;
    k_scatter<<<(unsigned)divup_ll(sc,256),256,0,stream>>>(xin, ei, ei + E, E, Cin, Kt,
                                                           idx4, w4, sbuf);
    long long ae = (long long)Mpad * Kpad;
    k_pack_A<<<(unsigned)divup_ll(ae,256),256,0,stream>>>(sbuf, xin, deg, Abuf,
                                                          nn, Cin, Kt, Kpad, Mpad);
    long long be = (long long)Npad * Kpad;
    k_pack_W<<<(unsigned)divup_ll(be,256),256,0,stream>>>(W, root, bias, Bbuf, biasp,
                                                          Cout, Cin, Kt, Kpad, Npad);
    dim3 g((unsigned)(Mpad / 32), (unsigned)(Npad / 32));
    k_gemm32<<<g, 32, 0, stream>>>(Abuf, Bbuf, biasp, xout, nn, Cout, Kpad, act);
  };

  auto pool = [&](const float* xin, float* xout, int nn, const int* cl,
                  const float* cnt, int nout, int C){
    zero(encb, (long long)nout * C);
    k_pool_max<<<(unsigned)divup_ll((long long)nn * C, 256),256,0,stream>>>(xin, cl, nn, C, encb);
    k_pool_fin<<<(unsigned)divup_ll((long long)nout * C, 256),256,0,stream>>>(encb, cnt, xout, nout, C);
  };

  auto gather = [&](const float* xin, float* xout, int nn, const int* cl, int C){
    k_gather<<<(unsigned)divup_ll((long long)nn * C, 256),256,0,stream>>>(xin, cl, xout, nn, C);
  };

  // ---------------- forward pass ----------------
  conv(x0,   bufX, n1, ei1, E1, idx15, w15, deg1,  6, 5,   1,  32, 1);  // c1a
  conv(bufX, bufY, n1, ei1, E1, idx15, w15, deg1,  9, 5,  32,  64, 1);  // c1b
  conv(bufY, bufX, n1, ei1, E1, idx15, w15, deg1, 12, 5,  64,  64, 1);  // c1c
  k_bn_stats<<<64, 256, 0, stream>>>(bufX, n1, 64, meanvar);
  k_bn_apply<<<(unsigned)divup_ll((long long)n1 * 64, 256),256,0,stream>>>(
      bufX, n1, 64, meanvar, bn_gamma, bn_beta);
  pool(bufX, bufY, n1, cl1, cnt1, n2, 64);                              // pool1
  conv(bufY, bufX, n2, ei2, E2, idx23, w23, deg2, 15, 3,  64, 128, 1);  // c2a
  conv(bufX, bufY, n2, ei2, E2, idx23, w23, deg2, 18, 3, 128, 128, 1);  // c2b
  conv(bufY, bufX, n2, ei2, E2, idx23, w23, deg2, 21, 3, 128, 256, 1);  // c2c
  pool(bufX, bufY, n2, cl2, cnt2, n2, 256);                             // pool2
  conv(bufY, bufX, n2, ei3, E3, idx33, w33, deg3, 24, 3, 256, 256, 1);  // c3a
  conv(bufX, bufY, n2, ei3, E3, idx33, w33, deg3, 27, 3, 256, 128, 1);  // c3b
  gather(bufY, bufX, n2, cl2, 128);                                     // unpool2
  conv(bufX, bufY, n2, ei2, E2, idx25, w25, deg2, 30, 5, 128,  64, 1);  // c4a
  conv(bufY, bufX, n2, ei2, E2, idx25, w25, deg2, 33, 5,  64,  32, 1);  // c4b
  gather(bufX, bufY, n1, cl1, 32);                                      // unpool1
  conv(bufY, (float*)d_out, n1, ei1, E1, idx15, w15, deg1, 36, 5, 32, 1, 2); // out+sigmoid
}
